// MSDeformAttnFuse_30734785970981
// MI455X (gfx1250) — matrix operand order repaired and checked by reading, not compiled, not run on hardware
//
#include <hip/hip_runtime.h>
#include <hip/hip_bf16.h>

// ---------------------------------------------------------------------------
// MS-deformable-attention fusion for gfx1250 (MI455X).
//   1. gemm<1>: value_img[M,Lq,D] = collab^T @ w_val + b_val
//   2. gemm<0>: off[Lq,64]        = ego^T @ w_off + b_off
//   3. gemm<0>: attn_logit[Lq,32] = ego^T @ w_attn + b_attn
//   4. softmax over P=4 groups
//   5. bilinear sample + head-weighted reduce -> tmp[C,Lq]
//   6. gemm<2>: out[C,Lq] = (tmp^T @ w_out + b_out)^T + ego  (residual)
// GEMMs: V_WMMA_F32_16X16X32_BF16, double-buffered LDS, branch-free staging,
// all shapes compile-time so address math folds to immediates.
// ---------------------------------------------------------------------------

typedef __attribute__((ext_vector_type(16))) __bf16 v16bf;
typedef __attribute__((ext_vector_type(8)))  __bf16 v8bf;
typedef __attribute__((ext_vector_type(8)))  float  v8f;

#define GEMM_K     256   // all GEMMs share K = C = 256
#define TILE_M     128
#define TILE_N     64
#define TILE_K     32
#define K_STEPS    (GEMM_K / TILE_K)     // 8
#define LDS_STRIDE 40    // bf16 elems/row: 80B, multiple of 16B

// Exact (lo, hi) -> packed-bf16 dword; inline asm defeats the SLP vectorizer's
// adjacent-element packing + bit-shuffle pattern.
__device__ __forceinline__ unsigned pack_bf16(float lo, float hi) {
    unsigned r;
    asm("v_cvt_pk_bf16_f32 %0, %1, %2" : "=v"(r) : "v"(lo), "v"(hi));
    return r;
}

// A is K-major: A[k*LQ + q].  Bw is [K, N] row-major.
// MODE 0: out[q*N + n]                     = acc + bias[n]
// MODE 1: out[((n>>5)*LQ + q)*32 + (n&31)] = acc + bias[n]   (value image [M,Lq,D])
// MODE 2: out[n*LQ + q]                    = acc + bias[n] + resid[n*LQ + q]
// Requires: LQ % 4 == 0, N % 4 == 0 (address-clamped staging).
template <int MODE, int N, int LQ>
__global__ __launch_bounds__(256)
void msda_gemm_bf16_wmma(const float* __restrict__ A,
                         const float* __restrict__ Bw,
                         const float* __restrict__ bias,
                         const float* __restrict__ resid,
                         float* __restrict__ out)
{
    __shared__ __bf16 sA[2][TILE_M][LDS_STRIDE];   // [m][k]
    __shared__ __bf16 sB[2][TILE_N][LDS_STRIDE];   // [n][k] (transposed)

    const int tid  = threadIdx.x;
    const int lane = tid & 31;            // wave32
    const int wave = tid >> 5;            // 0..7
    const int wm   = wave & 3;            // row group (4 x 32 rows)
    const int wn   = wave >> 2;           // col group (2 x 32 cols)

    const int q0 = blockIdx.x * TILE_M;
    const int n0 = blockIdx.y * TILE_N;

    // ---- branch-free staging coordinates (address-clamped, always aligned) ----
    const int a_q = (tid & 31) * 4;                       // 0,4,...,124
    const int a_k = (tid >> 5) * 2;                       // 0,2,...,14  (+16 on it=1)
    const int qg  = q0 + a_q;
    const int qc  = (qg + 3 < LQ) ? qg : (LQ - 4);        // 16B-aligned clamp
    const int b_n = (tid & 15) * 4;                       // 0,4,...,60
    const int b_k = (tid >> 4) * 2;                       // 0,2,...,30
    const int ng  = n0 + b_n;
    const int nc  = (ng + 3 < N) ? ng : (N - 4);          // 16B-aligned clamp

    // loop-invariant per-thread base pointers; per-stage offsets are constants
    const float* __restrict__ pA = A  + (size_t)a_k * LQ + qc;
    const float* __restrict__ pB = Bw + (size_t)b_k * N  + nc;

    auto stage = [&](int buf, int k0) {
        // A tile: 32k x 128q.  Two k-row pairs per thread, float4 along q.
        #pragma unroll
        for (int it = 0; it < 2; ++it) {
            const int k = a_k + it * 16;
            const float4 r0 = *(const float4*)(pA + (size_t)(k0 + it * 16) * LQ);
            const float4 r1 = *(const float4*)(pA + (size_t)(k0 + it * 16 + 1) * LQ);
            *(unsigned*)&sA[buf][a_q + 0][k] = pack_bf16(r0.x, r1.x);
            *(unsigned*)&sA[buf][a_q + 1][k] = pack_bf16(r0.y, r1.y);
            *(unsigned*)&sA[buf][a_q + 2][k] = pack_bf16(r0.z, r1.z);
            *(unsigned*)&sA[buf][a_q + 3][k] = pack_bf16(r0.w, r1.w);
        }
        // B tile: 32k x 64n.  One k-row pair per thread, float4 along n.
        {
            const float4 r0 = *(const float4*)(pB + (size_t)k0 * N);
            const float4 r1 = *(const float4*)(pB + (size_t)(k0 + 1) * N);
            *(unsigned*)&sB[buf][b_n + 0][b_k] = pack_bf16(r0.x, r1.x);
            *(unsigned*)&sB[buf][b_n + 1][b_k] = pack_bf16(r0.y, r1.y);
            *(unsigned*)&sB[buf][b_n + 2][b_k] = pack_bf16(r0.z, r1.z);
            *(unsigned*)&sB[buf][b_n + 3][b_k] = pack_bf16(r0.w, r1.w);
        }
        // L2 prefetch of the slab after next (global_prefetch_b8)
        if (k0 + 2 * TILE_K < GEMM_K)
            __builtin_prefetch(pA + (size_t)(k0 + 2 * TILE_K) * LQ, 0, 1);
    };

    v8f acc[2][2] = {};
    const int m16 = lane & 15;
    const int kh  = lane >> 4;            // K-half owned by this lane

    stage(0, 0);
    __syncthreads();

    #pragma unroll
    for (int kt = 0; kt < K_STEPS; ++kt) {
        const int cur = kt & 1;
        if (kt + 1 < K_STEPS)
            stage(cur ^ 1, (kt + 1) * TILE_K);   // overlap with WMMA on cur

        // ---- fragments per ISA 7.12.2 layouts ----
        v16bf afrag[2], bfrag[2];
        #pragma unroll
        for (int sm = 0; sm < 2; ++sm) {
            // A lane: K = kh*8..+7 and 16+kh*8..+7 at row M = m16
            const __bf16* pa = &sA[cur][wm * 32 + sm * 16 + m16][kh * 8];
            v8bf lo = *(const v8bf*)pa;
            v8bf hi = *(const v8bf*)(pa + 16);
            #pragma unroll
            for (int i = 0; i < 8; ++i) { afrag[sm][i] = lo[i]; afrag[sm][8 + i] = hi[i]; }
        }
        #pragma unroll
        for (int sn = 0; sn < 2; ++sn) {
            // B lane: K = kh*8..+7 and 16+kh*8..+7 at col N = m16
            const __bf16* pb = &sB[cur][wn * 32 + sn * 16 + m16][kh * 8];
            v8bf l0 = *(const v8bf*)pb;
            v8bf l1 = *(const v8bf*)(pb + 16);
            #pragma unroll
            for (int i = 0; i < 8; ++i) { bfrag[sn][i] = l0[i]; bfrag[sn][8 + i] = l1[i]; }
        }
        #pragma unroll
        for (int sm = 0; sm < 2; ++sm)
            #pragma unroll
            for (int sn = 0; sn < 2; ++sn)
                acc[sm][sn] = __builtin_amdgcn_wmma_f32_16x16x32_bf16(
                    false, afrag[sm], false, bfrag[sn],
                    (short)0, acc[sm][sn], false, false);

        __syncthreads();
    }

    // ---- epilogue: C/D layout (M = vgpr + 8*(lane>=16), N = lane&15) ----
    const int rowAdj = (lane >> 4) * 8;
    #pragma unroll
    for (int sm = 0; sm < 2; ++sm) {
        #pragma unroll
        for (int sn = 0; sn < 2; ++sn) {
            const int Rb = q0 + wm * 32 + sm * 16;
            const int Cb = n0 + wn * 32 + sn * 16;
            const int nn = Cb + (lane & 15);
            #pragma unroll
            for (int r = 0; r < 8; ++r) {
                const int mm = Rb + r + rowAdj;
                if (mm < LQ && nn < N) {
                    float v = acc[sm][sn][r] + bias[nn];
                    if (MODE == 0) {
                        out[(size_t)mm * N + nn] = v;
                    } else if (MODE == 1) {
                        out[((size_t)(nn >> 5) * LQ + mm) * 32 + (nn & 31)] = v;
                    } else {
                        size_t idx = (size_t)nn * LQ + mm;
                        out[idx] = v + resid[idx];
                    }
                }
            }
        }
    }
}

// softmax over the P=4 axis of attn logits [Lq, M, P]
__global__ __launch_bounds__(256)
void msda_softmax_p4(float* __restrict__ buf, int n_groups)
{
    int i = blockIdx.x * blockDim.x + threadIdx.x;   // (q*M + h)
    if (i >= n_groups) return;
    float4 x = *(float4*)(buf + (size_t)i * 4);
    float mx = fmaxf(fmaxf(x.x, x.y), fmaxf(x.z, x.w));
    float ea = __expf(x.x - mx), eb = __expf(x.y - mx);
    float ec = __expf(x.z - mx), ed = __expf(x.w - mx);
    float inv = 1.0f / (ea + eb + ec + ed);
    *(float4*)(buf + (size_t)i * 4) = make_float4(ea * inv, eb * inv, ec * inv, ed * inv);
}

// One block per query; wave = head h (8 waves), lane = channel d (D=32).
// Pixel coords simplify exactly: px = ix + off_x, py = iy + off_y.
// Writes tmp channel-major [C, Lq] so the final GEMM reuses the K-major path.
template <int H, int W>
__global__ __launch_bounds__(256)
void msda_sample(const float* __restrict__ val,   // [M, Lq, 32]
                 const float* __restrict__ off,   // [Lq, M*P*2]
                 const float* __restrict__ attn,  // [Lq, M*P]
                 float* __restrict__ tmp)         // [C, Lq]
{
    constexpr int LQ = H * W;
    const int q  = blockIdx.x;
    const int h  = threadIdx.x >> 5;
    const int d  = threadIdx.x & 31;
    const int ix = q % W;                 // constant divisor -> mul/shift
    const int iy = q / W;

    const float* vimg = val + (size_t)h * LQ * 32;
    float accum = 0.0f;

    #pragma unroll
    for (int p = 0; p < 4; ++p) {
        const float ox = off[(size_t)q * 64 + (size_t)(h * 4 + p) * 2 + 0];
        const float oy = off[(size_t)q * 64 + (size_t)(h * 4 + p) * 2 + 1];
        const float w  = attn[(size_t)q * 32 + h * 4 + p];
        const float px = (float)ix + ox;
        const float py = (float)iy + oy;
        const float x0f = floorf(px), y0f = floorf(py);
        const int   x0  = (int)x0f,  y0  = (int)y0f;
        const float wx1 = px - x0f,  wy1 = py - y0f;
        const float wx0 = 1.0f - wx1, wy0 = 1.0f - wy1;
        #pragma unroll
        for (int cy = 0; cy < 2; ++cy) {
            const int yy = y0 + cy;
            if (yy < 0 || yy >= H) continue;
            const float wy = cy ? wy1 : wy0;
            #pragma unroll
            for (int cx = 0; cx < 2; ++cx) {
                const int xx = x0 + cx;
                if (xx < 0 || xx >= W) continue;
                const float wx = cx ? wx1 : wx0;
                const float v = vimg[((size_t)yy * W + xx) * 32 + d];  // 128B coalesced
                accum = fmaf(v, wy * wx * w, accum);
            }
        }
    }
    tmp[(size_t)(h * 32 + d) * LQ + q] = accum;
}

extern "C" void kernel_launch(void* const* d_in, const int* in_sizes, int n_in,
                              void* d_out, int out_size, void* d_ws, size_t ws_size,
                              hipStream_t stream) {
    const float* ego    = (const float*)d_in[0];   // [C, Lq]
    const float* collab = (const float*)d_in[1];   // [C, Lq]
    const float* w_off  = (const float*)d_in[2];   // [256, 64]
    const float* b_off  = (const float*)d_in[3];
    const float* w_attn = (const float*)d_in[4];   // [256, 32]
    const float* b_attn = (const float*)d_in[5];
    const float* w_val  = (const float*)d_in[6];   // [256, 256]
    const float* b_val  = (const float*)d_in[7];
    const float* w_out  = (const float*)d_in[8];   // [256, 256]
    const float* b_out  = (const float*)d_in[9];
    float* out = (float*)d_out;                    // [C, H, W] = [C, Lq]

    constexpr int C = 256, H = 200, W = 200;
    constexpr int LQ = H * W;

    float* ws       = (float*)d_ws;
    float* val_img  = ws;                               // C*Lq  ([M,Lq,D])
    float* off_buf  = val_img  + (size_t)C * LQ;        // Lq*64
    float* attn_buf = off_buf  + (size_t)LQ * 64;       // Lq*32
    float* tmp      = attn_buf + (size_t)LQ * 32;       // C*Lq

    const dim3 blk(256);
    constexpr int mtiles = (LQ + TILE_M - 1) / TILE_M;  // 313

    // 1. value GEMM -> per-head value image [M, Lq, D]
    msda_gemm_bf16_wmma<1, 256, LQ><<<dim3(mtiles, 4), blk, 0, stream>>>(
        collab, w_val, b_val, nullptr, val_img);
    // 2. offsets GEMM -> [Lq, 64]
    msda_gemm_bf16_wmma<0, 64, LQ><<<dim3(mtiles, 1), blk, 0, stream>>>(
        ego, w_off, b_off, nullptr, off_buf);
    // 3. attention-logit GEMM -> [Lq, 32]
    msda_gemm_bf16_wmma<0, 32, LQ><<<dim3(mtiles, 1), blk, 0, stream>>>(
        ego, w_attn, b_attn, nullptr, attn_buf);
    // 4. softmax over P=4
    msda_softmax_p4<<<(LQ * 8 + 255) / 256, blk, 0, stream>>>(attn_buf, LQ * 8);
    // 5. bilinear sample + head reduce -> tmp [C, Lq]
    msda_sample<H, W><<<LQ, blk, 0, stream>>>(val_img, off_buf, attn_buf, tmp);
    // 6. output GEMM + bias + residual -> out [C, Lq]
    msda_gemm_bf16_wmma<2, 256, LQ><<<dim3(mtiles, 4), blk, 0, stream>>>(
        tmp, w_out, b_out, ego, out);
}
